// SWAttention_45449343926549
// MI455X (gfx1250) — matrix operand order repaired and checked by reading, not compiled, not run on hardware
//
#include <hip/hip_runtime.h>
#include <cmath>

// Problem constants (from reference)
#define BB   2
#define SS   2048
#define DM   2048
#define NH   32
#define DH   64
#define MAXB 1024

typedef __attribute__((ext_vector_type(16))) __bf16 v16bf;
typedef __attribute__((ext_vector_type(8)))  __bf16 v8bf;
typedef __attribute__((ext_vector_type(8)))  float  v8f;
typedef __attribute__((ext_vector_type(4)))  float  v4f;

// ---------- bf16 <-> f32 helpers (round-to-nearest-even) ----------
static __device__ __forceinline__ __bf16 f2bf(float f) {
  union { float f; unsigned u; } v; v.f = f;
  unsigned r = v.u + 0x7FFFu + ((v.u >> 16) & 1u);
  unsigned short h = (unsigned short)(r >> 16);
  __bf16 out;
  __builtin_memcpy(&out, &h, 2);
  return out;
}
static __device__ __forceinline__ float bf2f(__bf16 b) {
  unsigned short h;
  __builtin_memcpy(&h, &b, 2);
  union { unsigned u; float f; } v; v.u = ((unsigned)h) << 16;
  return v.f;
}

// ---------- WMMA fragment loaders (CDNA5 16x16x32 bf16 layouts) ----------
// A-matrix 16x32: lane holds row M=lane%16; vector elems 0..7 -> k = 8*half + i,
// elems 8..15 -> k = 16 + 8*half + (i-8). Two contiguous 16B chunks.
static __device__ __forceinline__ v16bf frag_a(const __bf16* row, int half) {
  v8bf lo = *(const v8bf*)(row + 8 * half);
  v8bf hi = *(const v8bf*)(row + 16 + 8 * half);
  v16bf r;
#pragma unroll
  for (int i = 0; i < 8; ++i) { r[i] = lo[i]; r[i + 8] = hi[i]; }
  return r;
}
// B-matrix 32x16: lane holds col N=lane%16; elems 0..7 -> k = 8*half + i,
// elems 8..15 -> k = 16 + 8*half + (i-8). Two contiguous 16B chunks.
static __device__ __forceinline__ v16bf frag_b(const __bf16* col, int half) {
  v8bf lo = *(const v8bf*)(col + 8 * half);
  v8bf hi = *(const v8bf*)(col + 16 + 8 * half);
  v16bf r;
#pragma unroll
  for (int i = 0; i < 8; ++i) { r[i] = lo[i]; r[i + 8] = hi[i]; }
  return r;
}

static __device__ __forceinline__ v8f wmma_bf16(v16bf a, v16bf b, v8f c) {
  return __builtin_amdgcn_wmma_f32_16x16x32_bf16(false, a, false, b, (short)0, c,
                                                 false, false);
}

// ---------- f32 -> bf16 convert (vectorized: 2x b128 load, 1x b128 store) ----
__global__ void cvt_bf16_kernel(const float* __restrict__ in,
                                __bf16* __restrict__ out, int n) {
  int i = (blockIdx.x * blockDim.x + threadIdx.x) * 8;
  const int stride = gridDim.x * blockDim.x * 8;
  for (; i < n; i += stride) {  // n is a multiple of 8 for all our tensors
    v4f a = *(const v4f*)(in + i);
    v4f b = *(const v4f*)(in + i + 4);
    v8bf o;
#pragma unroll
    for (int j = 0; j < 4; ++j) { o[j] = f2bf(a[j]); o[4 + j] = f2bf(b[j]); }
    *(v8bf*)(out + i) = o;
  }
}

// ---------- GEMM: C = A(MxK) @ W(NxK)^T, 64x64 register-blocked per wave ----
// 16 WMMAs per 32-deep k-step from 8 fragment loads (32 flops/byte operand reuse).
// mode 0: write bf16 to [B,H,S,D]      (Q / K projections)
// mode 2: write bf16 to [B,H,D,S]      (V projection, transposed)
// mode 3: write f32  row-major [M,N]   (output projection)
__global__ __launch_bounds__(128) void gemm_bf16_kernel(
    const __bf16* __restrict__ A, const __bf16* __restrict__ W,
    __bf16* __restrict__ outb, float* __restrict__ outf,
    int M, int N, int Kd, int mode) {
  const int wave = threadIdx.x >> 5;
  const int lane = threadIdx.x & 31;
  const int half = lane >> 4;
  const int ln   = lane & 15;
  const int tilesN = N >> 6;  // 64-wide tiles
  const int tile = blockIdx.x * 4 + wave;  // 4 consecutive N-tiles share A rows
  const int tm = tile / tilesN;
  const int tn = tile - tm * tilesN;
  if (tm * 64 >= M) return;

  const __bf16* arow[4];
  const __bf16* brow[4];
#pragma unroll
  for (int t = 0; t < 4; ++t) {
    arow[t] = A + (size_t)(tm * 64 + t * 16 + ln) * Kd;
    brow[t] = W + (size_t)(tn * 64 + t * 16 + ln) * Kd;
  }
  v8f acc[4][4] = {};
#pragma unroll 1
  for (int k = 0; k < Kd; k += 32) {
    v16bf af[4], bf[4];
#pragma unroll
    for (int t = 0; t < 4; ++t) {
      af[t] = frag_a(arow[t] + k, half);
      bf[t] = frag_b(brow[t] + k, half);
      __builtin_prefetch(arow[t] + k + 256, 0, 3);  // near-scope prefetch, 8 k-steps ahead
      __builtin_prefetch(brow[t] + k + 256, 0, 3);
    }
#pragma unroll
    for (int mi = 0; mi < 4; ++mi)
#pragma unroll
      for (int ni = 0; ni < 4; ++ni)
        acc[mi][ni] = wmma_bf16(af[mi], bf[ni], acc[mi][ni]);
  }

#pragma unroll
  for (int mi = 0; mi < 4; ++mi) {
#pragma unroll
    for (int ni = 0; ni < 4; ++ni) {
      const int gn = tn * 64 + ni * 16 + ln;
#pragma unroll
      for (int r = 0; r < 8; ++r) {
        const int gm = tm * 64 + mi * 16 + r + 8 * half;  // C layout: lane=col, VGPR r=row
        const float v = acc[mi][ni][r];
        if (mode == 3) {
          outf[(size_t)gm * N + gn] = v;
        } else {
          const int b = gm / SS, s = gm - b * SS;
          const int h = gn >> 6, d = gn & 63;
          size_t off;
          if (mode == 2) off = (((size_t)b * NH + h) * DH + d) * SS + s;  // V^T
          else           off = (((size_t)b * NH + h) * SS + s) * DH + d;  // Q/K
          outb[off] = f2bf(v);
        }
      }
    }
  }
}

// ---------- RoPE applied in place to Q and K ([B*H, S, D] bf16) ----------
__global__ void rope_kernel(__bf16* __restrict__ Q, __bf16* __restrict__ K) {
  const int idx = blockIdx.x * blockDim.x + threadIdx.x;  // (bh, s, d<32)
  const int total = BB * NH * SS * 32;
  if (idx >= total) return;
  const int d  = idx & 31;
  const int s  = (idx >> 5) & (SS - 1);
  const int bh = idx >> 16;
  // inv_freq = 10000^(-d/32) = exp(-d * ln(10000)/32)
  const float inv = __expf(-(float)d * (9.210340372f / 32.0f));
  const float ang = (float)s * inv;
  const float c = cosf(ang), sn = sinf(ang);
  const size_t base = ((size_t)bh * SS + s) * DH + d;
  {
    const float x1 = bf2f(Q[base]), x2 = bf2f(Q[base + 32]);
    Q[base]      = f2bf(x1 * c - x2 * sn);
    Q[base + 32] = f2bf(x1 * sn + x2 * c);
  }
  {
    const float x1 = bf2f(K[base]), x2 = bf2f(K[base + 32]);
    K[base]      = f2bf(x1 * c - x2 * sn);
    K[base + 32] = f2bf(x1 * sn + x2 * c);
  }
}

// ---------- Flash attention: 1 wave per 16 query rows, 32-wide KV blocks ----
__global__ __launch_bounds__(128) void flash_kernel(
    const __bf16* __restrict__ Q,   // [B*H, S, D]
    const __bf16* __restrict__ K,   // [B*H, S, D]
    const __bf16* __restrict__ Vt,  // [B*H, D, S]
    const float* __restrict__ lb,   // [H, MAXB]
    const float* __restrict__ soff, // [H]
    __bf16* __restrict__ O) {       // [B, S, Dm]
  __shared__ __align__(16) __bf16 ldsP[4][16][32];  // per-wave P staging
  const int wave = threadIdx.x >> 5;
  const int lane = threadIdx.x & 31;
  const int half = lane >> 4;
  const int ln   = lane & 15;
  const int g  = blockIdx.x * 4 + wave;
  const int qt = g & 127;   // S/16 = 128 query tiles
  const int bh = g >> 7;    // [0, B*H)
  const int h  = bh & (NH - 1);
  const float scale = 0.125f;  // 64^-0.5
  const float off = soff[h];

  // Q fragments, loaded once (K-dim = 64 -> two 32-deep steps)
  const __bf16* qrow = Q + ((size_t)bh * SS + qt * 16 + ln) * DH;
  const v16bf qa0 = frag_a(qrow, half);
  const v16bf qa1 = frag_a(qrow + 32, half);

  float mrow[8], lrow[8];
  v8f o0 = {}, o1 = {}, o2 = {}, o3 = {};
#pragma unroll
  for (int r = 0; r < 8; ++r) { mrow[r] = off; lrow[r] = 0.0f; }

  const int iEnd = (qt + 1) * 16;
  for (int jb = 0; jb < iEnd; jb += 32) {
    float p0[8], p1[8];
    // -------- scores: two 16x16 tiles (QK^T), scale + bias + causal mask ----
#pragma unroll
    for (int jt = 0; jt < 2; ++jt) {
      const int j0 = jb + jt * 16;
      const __bf16* krow = K + ((size_t)bh * SS + j0 + ln) * DH;
      v8f sacc = {};
      sacc = wmma_bf16(qa0, frag_b(krow, half), sacc);
      sacc = wmma_bf16(qa1, frag_b(krow + 32, half), sacc);
      const int j = j0 + ln;
      float* p = jt ? p1 : p0;
#pragma unroll
      for (int r = 0; r < 8; ++r) {
        const int i = qt * 16 + r + 8 * half;
        const int diag = i - j;
        float sc = sacc[r] * scale;
        if (diag >= 0 && diag < MAXB) sc += lb[h * MAXB + diag];
        if (diag < 0) sc = -INFINITY;
        p[r] = sc;
      }
    }
    // -------- online softmax: row reductions across the 16-lane half --------
#pragma unroll
    for (int r = 0; r < 8; ++r) {
      float mx = fmaxf(p0[r], p1[r]);
      mx = fmaxf(mx, __shfl_xor(mx, 1));
      mx = fmaxf(mx, __shfl_xor(mx, 2));
      mx = fmaxf(mx, __shfl_xor(mx, 4));
      mx = fmaxf(mx, __shfl_xor(mx, 8));
      const float mnew  = fmaxf(mrow[r], mx);
      const float alpha = __expf(mrow[r] - mnew);
      p0[r] = __expf(p0[r] - mnew);
      p1[r] = __expf(p1[r] - mnew);
      float sum = p0[r] + p1[r];
      sum += __shfl_xor(sum, 1);
      sum += __shfl_xor(sum, 2);
      sum += __shfl_xor(sum, 4);
      sum += __shfl_xor(sum, 8);
      lrow[r] = lrow[r] * alpha + sum;
      mrow[r] = mnew;
      o0[r] *= alpha; o1[r] *= alpha; o2[r] *= alpha; o3[r] *= alpha;
      // stage P (C layout -> LDS) for A-layout reload
      ldsP[wave][r + 8 * half][ln]      = f2bf(p0[r]);
      ldsP[wave][r + 8 * half][16 + ln] = f2bf(p1[r]);
    }
    // -------- P·V: reload P as A-fragment, V^T columns are contiguous -------
    const v16bf pa = frag_a(&ldsP[wave][ln][0], half);
#pragma unroll
    for (int t = 0; t < 4; ++t) {
      const __bf16* vcol = Vt + ((size_t)bh * DH + t * 16 + ln) * SS + jb;
      v8f& oacc = (t == 0) ? o0 : (t == 1) ? o1 : (t == 2) ? o2 : o3;
      oacc = wmma_bf16(pa, frag_b(vcol, half), oacc);
    }
  }
  // -------- epilogue: divide by (l + exp(off - m)), write bf16 [B,S,Dm] -----
  const int b = bh >> 5;  // NH == 32
#pragma unroll
  for (int r = 0; r < 8; ++r) {
    const float denom = lrow[r] + __expf(off - mrow[r]);
    const float inv = 1.0f / denom;
    const int i = qt * 16 + r + 8 * half;
    const size_t rowbase = ((size_t)b * SS + i) * DM + (size_t)h * DH;
    O[rowbase + ln]      = f2bf(o0[r] * inv);
    O[rowbase + 16 + ln] = f2bf(o1[r] * inv);
    O[rowbase + 32 + ln] = f2bf(o2[r] * inv);
    O[rowbase + 48 + ln] = f2bf(o3[r] * inv);
  }
}

extern "C" void kernel_launch(void* const* d_in, const int* in_sizes, int n_in,
                              void* d_out, int out_size, void* d_ws, size_t ws_size,
                              hipStream_t stream) {
  (void)in_sizes; (void)n_in; (void)out_size; (void)ws_size;
  const float* hx = (const float*)d_in[0];
  const float* wq = (const float*)d_in[1];
  const float* wk = (const float*)d_in[2];
  const float* wv = (const float*)d_in[3];
  const float* wo = (const float*)d_in[4];
  const float* lb = (const float*)d_in[5];
  const float* so = (const float*)d_in[6];
  float* out = (float*)d_out;

  const size_t nX = (size_t)BB * SS * DM;  // 8,388,608
  const size_t nW = (size_t)DM * DM;       // 4,194,304

  char* ws = (char*)d_ws;
  __bf16* Xb  = (__bf16*)ws;                 ws += nX * 2;
  __bf16* Wqb = (__bf16*)ws;                 ws += nW * 2;
  __bf16* Wkb = (__bf16*)ws;                 ws += nW * 2;
  __bf16* Wvb = (__bf16*)ws;                 ws += nW * 2;
  __bf16* Wob = (__bf16*)ws;                 ws += nW * 2;
  __bf16* Qb  = (__bf16*)ws;                 ws += nX * 2;
  __bf16* Kb  = (__bf16*)ws;                 ws += nX * 2;
  __bf16* Vtb = (__bf16*)ws;                 ws += nX * 2;
  __bf16* Ob  = (__bf16*)ws;                 ws += nX * 2;  // ~112 MB total

  // 1) convert inputs to bf16 (8 elems/thread, b128 stores)
  cvt_bf16_kernel<<<1024, 256, 0, stream>>>(hx, Xb, (int)nX);
  cvt_bf16_kernel<<<512, 256, 0, stream>>>(wq, Wqb, (int)nW);
  cvt_bf16_kernel<<<512, 256, 0, stream>>>(wk, Wkb, (int)nW);
  cvt_bf16_kernel<<<512, 256, 0, stream>>>(wv, Wvb, (int)nW);
  cvt_bf16_kernel<<<512, 256, 0, stream>>>(wo, Wob, (int)nW);

  // 2) projections (WMMA GEMMs), 64x64 tile per wave, 4 waves per block
  const int M = BB * SS, N = DM, Kd = DM;
  const int blocks = (M / 64) * (N / 64) / 4;  // 512
  gemm_bf16_kernel<<<blocks, 128, 0, stream>>>(Xb, Wqb, Qb, nullptr, M, N, Kd, 0);
  gemm_bf16_kernel<<<blocks, 128, 0, stream>>>(Xb, Wkb, Kb, nullptr, M, N, Kd, 0);
  gemm_bf16_kernel<<<blocks, 128, 0, stream>>>(Xb, Wvb, Vtb, nullptr, M, N, Kd, 2);

  // 3) RoPE on Q and K
  rope_kernel<<<(BB * NH * SS * 32) / 256, 256, 0, stream>>>(Qb, Kb);

  // 4) causal flash attention with banded bias + offset softmax
  flash_kernel<<<(BB * NH * (SS / 16)) / 4, 128, 0, stream>>>(Qb, Kb, Vtb, lb, so, Ob);

  // 5) output projection -> f32
  gemm_bf16_kernel<<<blocks, 128, 0, stream>>>(Ob, Wob, nullptr, out, M, N, Kd, 3);
}
